// TrajectoryAttention_30709016167161
// MI455X (gfx1250) — matrix operand order repaired and checked by reading, not compiled, not run on hardware
//
#include <hip/hip_runtime.h>

// ---------------------------------------------------------------------------
// Types / constants
// ---------------------------------------------------------------------------
typedef __attribute__((ext_vector_type(16))) __bf16 v16bf;
typedef __attribute__((ext_vector_type(8)))  __bf16 v8bf;
typedef __attribute__((ext_vector_type(8)))  float  v8f;
typedef __attribute__((ext_vector_type(4)))  unsigned int u32x4;
typedef __attribute__((ext_vector_type(4)))  int i32x4;
typedef __attribute__((ext_vector_type(8)))  int i32x8;
typedef unsigned short u16t;

static constexpr int Bc  = 2;
static constexpr int Tc  = 10;
static constexpr int Nc  = 1024;
static constexpr int Dc  = 1024;
static constexpr int NHc = 16;
static constexpr int HDc = 64;
static constexpr int Mrows = Bc * Nc;   // 2048 GEMM rows

// float -> bf16 round-to-nearest-even
__device__ __forceinline__ u16t f2bf(float f) {
  unsigned int u = __float_as_uint(f);
  u += 0x7fffu + ((u >> 16) & 1u);
  return (u16t)(u >> 16);
}

// ---------------------------------------------------------------------------
// WMMA fragment helpers (wave32; A and B share contiguous k order k = 16*(lane>=16) + e)
// ---------------------------------------------------------------------------
// A-matrix 16x32 bf16: lane row = lane%16; halves 0-15 = K koff..koff+15,
// koff = 16*(lane>=16).
__device__ __forceinline__ v16bf load_frag_a(const u16t* base, int ld, int lane) {
  int row  = lane & 15;
  int sel8 = (lane >> 4) << 4;
  const u16t* p = base + row * ld + sel8;
  v8bf lo = *(const v8bf*)(p);
  v8bf hi = *(const v8bf*)(p + 8);
  return __builtin_shufflevector(lo, hi, 0,1,2,3,4,5,6,7,8,9,10,11,12,13,14,15);
}

// B-matrix 32x16 bf16, B[k][n] with source rows = n, contiguous in k
// (i.e. column-major B == row-major W). lane col = lane%16,
// 16 contiguous K halves at koff = 16*(lane>=16).
__device__ __forceinline__ v16bf load_frag_b(const u16t* base, int ld, int lane) {
  int n    = lane & 15;
  int koff = (lane >> 4) << 4;
  return *(const v16bf*)(base + n * ld + koff);
}

// A fragment assembled from f32 data in LDS (convert to bf16 per element)
__device__ __forceinline__ v16bf load_frag_a_f32(const float* Sbase, int ld, int k0, int lane) {
  int row  = lane & 15;
  int sel8 = (lane >> 4) << 4;
  const float* p = Sbase + row * ld + k0 + sel8;
  union { v16bf v; u16t s[16]; } u;
#pragma unroll
  for (int e = 0; e < 8; ++e) u.s[e]     = f2bf(p[e]);
#pragma unroll
  for (int e = 0; e < 8; ++e) u.s[e + 8] = f2bf(p[8 + e]);
  return u.v;
}

__device__ __forceinline__ v8f wmma_bf16(v16bf a, v16bf b, v8f c) {
  return __builtin_amdgcn_wmma_f32_16x16x32_bf16(
      /*neg_a=*/false, a, /*neg_b=*/false, b,
      /*c_mod=*/(short)0, c, /*reuse_a=*/false, /*reuse_b=*/false);
}

// ---------------------------------------------------------------------------
// Tensor Data Mover: load a 2D bf16 tile (tile_k x tile_rows, row stride
// row_stride elements) from global into LDS (compacted row-major).
// D# layout per CDNA5 ISA 8.3/8.4 (group0: count/lds/global/type,
// group1: data_size=2B, tensor dims, tile dims, dim0 stride).
// ---------------------------------------------------------------------------
__device__ __forceinline__ void tdm_load_tile_bf16(unsigned lds_off, const u16t* gsrc,
                                                   int tile_k, int tile_rows,
                                                   int tensor_d0, int tensor_d1,
                                                   int row_stride) {
  unsigned long long ga = (unsigned long long)(size_t)gsrc;
  u32x4 g0;
  g0[0] = 1u;                                             // count=1, user mode
  g0[1] = lds_off;                                        // lds_addr (bytes)
  g0[2] = (unsigned)(ga & 0xffffffffu);                   // global_addr[31:0]
  g0[3] = (unsigned)((ga >> 32) & 0x01ffffffu) | (2u << 30); // addr[56:32] | type=2
  i32x8 g1;
  g1[0] = 0x00010000;                                     // data_size=1 (2 bytes)
  g1[1] = (int)(((unsigned)tensor_d0 & 0xffffu) << 16);   // tensor_dim0[15:0]
  g1[2] = (int)(((unsigned)tensor_d0 >> 16) |
                (((unsigned)tensor_d1 & 0xffffu) << 16)); // dim0[31:16] | dim1[15:0]
  g1[3] = (int)(((unsigned)tensor_d1 >> 16) |
                ((unsigned)tile_k << 16));                // dim1[31:16] | tile_dim0
  g1[4] = tile_rows;                                      // tile_dim1 (tile_dim2=0)
  g1[5] = row_stride;                                     // tensor_dim0_stride[31:0]
  g1[6] = 0;                                              // stride hi / dim1_stride lo
  g1[7] = 0;
  i32x4 z4 = {0, 0, 0, 0};
#if defined(__clang_major__) && (__clang_major__ >= 23)
  i32x8 z8 = {0, 0, 0, 0, 0, 0, 0, 0};
  __builtin_amdgcn_tensor_load_to_lds(g0, g1, z4, z4, z8, 0);
#else
  __builtin_amdgcn_tensor_load_to_lds(g0, g1, z4, z4, 0);
#endif
}

// ---------------------------------------------------------------------------
// f32 -> bf16 conversion
// ---------------------------------------------------------------------------
__global__ void cvt_bf16_kernel(const float* __restrict__ in, u16t* __restrict__ out, int n) {
  for (int i = blockIdx.x * blockDim.x + threadIdx.x; i < n; i += gridDim.x * blockDim.x)
    out[i] = f2bf(in[i]);
}

// ---------------------------------------------------------------------------
// Trajectory similarity: traj[i][j] row-normalized exp(-pd - min(qdm,qdp))
// One block per row i, 256 threads over j.
// ---------------------------------------------------------------------------
__global__ __launch_bounds__(256) void traj_kernel(const float* __restrict__ pos,
                                                   const float* __restrict__ quat,
                                                   float* __restrict__ traj) {
  const int i = blockIdx.x;
  const int t = threadIdx.x;
  __shared__ float pi[Bc * Tc * 3];
  __shared__ float qi[Bc * Tc * 4];
  __shared__ float red[256];
  if (t < Bc * Tc * 3) pi[t] = pos[(t / 3) * (Nc * 3) + i * 3 + (t % 3)];
  if (t < Bc * Tc * 4) qi[t] = quat[(t / 4) * (Nc * 4) + i * 4 + (t % 4)];
  __syncthreads();

  float vals[4];
#pragma unroll
  for (int jj = 0; jj < 4; ++jj) {
    const int j = t + jj * 256;
    float pd = 0.f, qm = 0.f, qp = 0.f;
    for (int bt = 0; bt < Bc * Tc; ++bt) {
      const float* pj = pos + (bt * Nc + j) * 3;
      float dx = pi[bt * 3 + 0] - pj[0];
      float dy = pi[bt * 3 + 1] - pj[1];
      float dz = pi[bt * 3 + 2] - pj[2];
      pd += sqrtf(dx * dx + dy * dy + dz * dz);
      const float* qj = quat + (bt * Nc + j) * 4;
      float m0 = qi[bt * 4 + 0] - qj[0], p0 = qi[bt * 4 + 0] + qj[0];
      float m1 = qi[bt * 4 + 1] - qj[1], p1 = qi[bt * 4 + 1] + qj[1];
      float m2 = qi[bt * 4 + 2] - qj[2], p2 = qi[bt * 4 + 2] + qj[2];
      float m3 = qi[bt * 4 + 3] - qj[3], p3 = qi[bt * 4 + 3] + qj[3];
      qm += sqrtf(m0 * m0 + m1 * m1 + m2 * m2 + m3 * m3);
      qp += sqrtf(p0 * p0 + p1 * p1 + p2 * p2 + p3 * p3);
    }
    pd *= 0.05f; qm *= 0.05f; qp *= 0.05f;
    vals[jj] = expf(-(pd + fminf(qm, qp)));
  }
  red[t] = vals[0] + vals[1] + vals[2] + vals[3];
  __syncthreads();
  for (int off = 128; off > 0; off >>= 1) {
    if (t < off) red[t] += red[t + off];
    __syncthreads();
  }
  const float inv = 1.0f / red[0];
#pragma unroll
  for (int jj = 0; jj < 4; ++jj)
    traj[i * Nc + t + jj * 256] = vals[jj] * inv;
}

// ---------------------------------------------------------------------------
// bf16 WMMA GEMM:  out[m][n] = sum_k A[m][k] * W[n][k] + bias[n]
// Both A (64x128) and W (128x128) tiles staged in LDS by the Tensor Data
// Mover, double-buffered, wave 0 drives the DMA (2 descriptors per stage,
// pipelined via s_wait_tensorcnt(2)). 8 stages of 128 K; between barriers
// each wave issues 16 WMMAs fed purely by ds_load_b128.
// MODE 0: bf16 row-major out    (Q, K projections)
// MODE 1: bf16 V-transposed out (b,h,hd,n) layout for P*V B-fragments
// MODE 2: f32 row-major out     (final Wo projection -> d_out)
// Block 256 threads (8 waves), block tile 64x128, wave tile 32x32.
// ---------------------------------------------------------------------------
template <int MODE>
__global__ __launch_bounds__(256) void gemm_bf16_kernel(const u16t* __restrict__ A,
                                                        const u16t* __restrict__ W,
                                                        const float* __restrict__ bias,
                                                        void* __restrict__ out) {
  constexpr int KSTAGE = 128;
  constexpr int NSTAGE = Dc / KSTAGE;      // 8 stages

  const int lane = threadIdx.x & 31;
  const int wave = threadIdx.x >> 5;
  const int bm = blockIdx.x * 64;
  const int bn = blockIdx.y * 128;
  const int m0 = bm + (wave >> 2) * 32;
  const int n0 = bn + (wave & 3) * 32;
  const int lm = (wave >> 2) * 32;         // wave row offset inside A tile
  const int ln = (wave & 3) * 32;          // wave col offset inside W tile

  __shared__ u16t tileA[2][64 * KSTAGE];   // 2 x 16 KB
  __shared__ u16t tileW[2][128 * KSTAGE];  // 2 x 32 KB

  // Prologue: DMA stage 0 (A tile + W tile).
  if (wave == 0) {
    tdm_load_tile_bf16((unsigned)(size_t)&tileA[0][0], A + (size_t)bm * Dc,
                       KSTAGE, 64, Dc, Mrows, Dc);
    tdm_load_tile_bf16((unsigned)(size_t)&tileW[0][0], W + (size_t)bn * Dc,
                       KSTAGE, 128, Dc, Dc, Dc);
  }

  v8f c[2][2] = {};
  for (int s = 0; s < NSTAGE; ++s) {
    if (wave == 0) {
      if (s + 1 < NSTAGE) {
        tdm_load_tile_bf16((unsigned)(size_t)&tileA[(s + 1) & 1][0],
                           A + (size_t)bm * Dc + (s + 1) * KSTAGE,
                           KSTAGE, 64, Dc, Mrows, Dc);
        tdm_load_tile_bf16((unsigned)(size_t)&tileW[(s + 1) & 1][0],
                           W + (size_t)bn * Dc + (s + 1) * KSTAGE,
                           KSTAGE, 128, Dc, Dc, Dc);
        __builtin_amdgcn_s_wait_tensorcnt(2);  // stage s's two loads complete
      } else {
        __builtin_amdgcn_s_wait_tensorcnt(0);
      }
    }
    __syncthreads();                       // publish stage s to all waves

    const u16t* At = &tileA[s & 1][0];
    const u16t* Wt = &tileW[s & 1][0];
#pragma unroll
    for (int kk = 0; kk < KSTAGE; kk += 32) {
      v16bf a0 = load_frag_a(At + (lm +  0) * KSTAGE + kk, KSTAGE, lane);
      v16bf a1 = load_frag_a(At + (lm + 16) * KSTAGE + kk, KSTAGE, lane);
      v16bf b0 = load_frag_b(Wt + (ln +  0) * KSTAGE + kk, KSTAGE, lane);
      v16bf b1 = load_frag_b(Wt + (ln + 16) * KSTAGE + kk, KSTAGE, lane);
      c[0][0] = wmma_bf16(a0, b0, c[0][0]);
      c[0][1] = wmma_bf16(a0, b1, c[0][1]);
      c[1][0] = wmma_bf16(a1, b0, c[1][0]);
      c[1][1] = wmma_bf16(a1, b1, c[1][1]);
    }

    __syncthreads();                       // stage buffer free for overwrite
  }

  const int colb = lane & 15;
  const int rofs = (lane >> 4) << 3;
#pragma unroll
  for (int mi = 0; mi < 2; ++mi)
#pragma unroll
    for (int ni = 0; ni < 2; ++ni)
#pragma unroll
      for (int r = 0; r < 8; ++r) {
        const int row = m0 + mi * 16 + rofs + r;   // global M index (b*N+n)
        const int col = n0 + ni * 16 + colb;       // global N index (d)
        const float v = c[mi][ni][r] + bias[col];
        if (MODE == 0) {
          ((u16t*)out)[row * Dc + col] = f2bf(v);
        } else if (MODE == 1) {
          const int bidx = row >> 10, n = row & (Nc - 1);
          const int h = col >> 6, hd = col & (HDc - 1);
          ((u16t*)out)[(((bidx * NHc) + h) * HDc + hd) * Nc + n] = f2bf(v);
        } else {
          ((float*)out)[row * Dc + col] = v;
        }
      }
}

// ---------------------------------------------------------------------------
// Fused attention: one block per (b, 16-row tile), 128 threads (4 waves),
// loops over all 16 heads. S strip (16x1024 f32) lives in LDS (64 KB).
// Entropy of head-averaged attention accumulated in registers.
// ---------------------------------------------------------------------------
__global__ __launch_bounds__(128) void attn_kernel(const u16t* __restrict__ Qp,
                                                   const u16t* __restrict__ Kp,
                                                   const u16t* __restrict__ Vt,
                                                   const float* __restrict__ traj,
                                                   const float* __restrict__ cert_in,
                                                   u16t* __restrict__ attn_out,
                                                   float* __restrict__ cert_out) {
  const int b  = blockIdx.x >> 6;
  const int i0 = (blockIdx.x & 63) << 4;
  const int t    = threadIdx.x;
  const int lane = t & 31;
  const int wave = t >> 5;

  __shared__ float S[16 * Nc];   // 64 KB score/probability strip

  // softmax / entropy ownership: row = t>>3, cols = (t&7) + 8*k, k = 0..127
  const int erow  = t >> 3;
  const int ecol0 = t & 7;
  float pacc[128];
#pragma unroll
  for (int k = 0; k < 128; ++k) pacc[k] = 0.f;

  for (int h = 0; h < NHc; ++h) {
    // ---- Phase 1: S = Q K^T / sqrt(HD) + 0.5 * traj -----------------------
    v16bf qa0 = load_frag_a(Qp + (b * Nc + i0) * Dc + h * HDc +  0, Dc, lane);
    v16bf qa1 = load_frag_a(Qp + (b * Nc + i0) * Dc + h * HDc + 32, Dc, lane);
    for (int jt = wave; jt < 64; jt += 4) {
      const int j0 = jt * 16;
      v8f c = {};
      v16bf kb0 = load_frag_b(Kp + (b * Nc + j0) * Dc + h * HDc +  0, Dc, lane);
      v16bf kb1 = load_frag_b(Kp + (b * Nc + j0) * Dc + h * HDc + 32, Dc, lane);
      c = wmma_bf16(qa0, kb0, c);
      c = wmma_bf16(qa1, kb1, c);
      const int colb = lane & 15;
      const int rofs = (lane >> 4) << 3;
#pragma unroll
      for (int r = 0; r < 8; ++r) {
        const int m = rofs + r;
        const int col = j0 + colb;
        S[m * Nc + col] = c[r] * 0.125f + 0.5f * traj[(i0 + m) * Nc + col];
      }
    }
    __syncthreads();

    // ---- Softmax over each row (barrier-free; 8 lanes per row) ------------
    float mx = -1e30f;
#pragma unroll
    for (int k = 0; k < 128; ++k)
      mx = fmaxf(mx, S[erow * Nc + ecol0 + k * 8]);
    mx = fmaxf(mx, __shfl_xor(mx, 1, 32));
    mx = fmaxf(mx, __shfl_xor(mx, 2, 32));
    mx = fmaxf(mx, __shfl_xor(mx, 4, 32));

    float sum = 0.f;
#pragma unroll
    for (int k = 0; k < 128; ++k) {
      const int idx = erow * Nc + ecol0 + k * 8;
      const float e = expf(S[idx] - mx);
      S[idx] = e;
      sum += e;
    }
    sum += __shfl_xor(sum, 1, 32);
    sum += __shfl_xor(sum, 2, 32);
    sum += __shfl_xor(sum, 4, 32);
    const float inv = 1.0f / sum;
#pragma unroll
    for (int k = 0; k < 128; ++k) {
      const int idx = erow * Nc + ecol0 + k * 8;
      const float p = S[idx] * inv;
      S[idx] = p;
      pacc[k] += p * (1.0f / (float)NHc);   // head-averaged attention
    }
    __syncthreads();

    // ---- Phase 2: O = P V (P from LDS, V transposed in ws) ----------------
    const int hd0 = wave * 16;
    v8f c2 = {};
    const u16t* vb = Vt + ((b * NHc + h) * HDc + hd0 + (lane & 15)) * Nc
                        + ((lane >> 4) << 4);
#pragma unroll 4
    for (int k0 = 0; k0 < Nc; k0 += 32) {
      v16bf a  = load_frag_a_f32(S, Nc, k0, lane);
      v16bf bb = *(const v16bf*)(vb + k0);
      c2 = wmma_bf16(a, bb, c2);
    }
    const int colb = lane & 15;
    const int rofs = (lane >> 4) << 3;
#pragma unroll
    for (int r = 0; r < 8; ++r) {
      const int m = rofs + r;
      const int col = h * HDc + hd0 + colb;
      attn_out[(b * Nc + i0 + m) * Dc + col] = f2bf(c2[r]);
    }
    __syncthreads();   // protect S before next head's phase 1
  }

  // ---- Entropy of avg attention -> certainty update ----------------------
  float Hp = 0.f;
#pragma unroll
  for (int k = 0; k < 128; ++k) {
    const float p = fmaxf(pacc[k], 1e-10f);
    Hp -= p * logf(p);
  }
  Hp += __shfl_xor(Hp, 1, 32);
  Hp += __shfl_xor(Hp, 2, 32);
  Hp += __shfl_xor(Hp, 4, 32);
  if ((t & 7) == 0) {
    const float Hmax = 6.931471805599453f;  // log(1024)
    const float x = Hmax - Hp;
    const float sg = 1.0f / (1.0f + expf(-x));
    const int gi = b * Nc + i0 + erow;
    cert_out[gi] = fmaxf(cert_in[gi], sg);
  }
}

// ---------------------------------------------------------------------------
// Launch
// ---------------------------------------------------------------------------
extern "C" void kernel_launch(void* const* d_in, const int* in_sizes, int n_in,
                              void* d_out, int out_size, void* d_ws, size_t ws_size,
                              hipStream_t stream) {
  const float* data  = (const float*)d_in[0];
  const float* tpos  = (const float*)d_in[1];
  const float* tquat = (const float*)d_in[2];
  const float* cert  = (const float*)d_in[3];
  const float* Wq = (const float*)d_in[4];  const float* bq = (const float*)d_in[5];
  const float* Wk = (const float*)d_in[6];  const float* bk = (const float*)d_in[7];
  const float* Wv = (const float*)d_in[8];  const float* bv = (const float*)d_in[9];
  const float* Wo = (const float*)d_in[10]; const float* bo = (const float*)d_in[11];

  // Workspace carve-up (all sizes 256B aligned)
  char* w = (char*)d_ws;
  u16t* data16 = (u16t*)w;  w += (size_t)Mrows * Dc * 2;       // 4 MB
  u16t* wq16   = (u16t*)w;  w += (size_t)Dc * Dc * 2;          // 2 MB
  u16t* wk16   = (u16t*)w;  w += (size_t)Dc * Dc * 2;
  u16t* wv16   = (u16t*)w;  w += (size_t)Dc * Dc * 2;
  u16t* wo16   = (u16t*)w;  w += (size_t)Dc * Dc * 2;
  u16t* Qp     = (u16t*)w;  w += (size_t)Mrows * Dc * 2;       // 4 MB
  u16t* Kp     = (u16t*)w;  w += (size_t)Mrows * Dc * 2;
  u16t* Vt     = (u16t*)w;  w += (size_t)Mrows * Dc * 2;       // (b,h,hd,n)
  float* traj  = (float*)w; w += (size_t)Nc * Nc * 4;          // 4 MB
  u16t* aout   = (u16t*)w;  w += (size_t)Mrows * Dc * 2;       // 4 MB

  float* out_f  = (float*)d_out;                 // (B,N,D)
  float* cert_o = out_f + (size_t)Bc * Nc * Dc;  // (B,N) tail

  // 1) bf16 casts
  cvt_bf16_kernel<<<2048, 256, 0, stream>>>(data, data16, Mrows * Dc);
  cvt_bf16_kernel<<<1024, 256, 0, stream>>>(Wq, wq16, Dc * Dc);
  cvt_bf16_kernel<<<1024, 256, 0, stream>>>(Wk, wk16, Dc * Dc);
  cvt_bf16_kernel<<<1024, 256, 0, stream>>>(Wv, wv16, Dc * Dc);
  cvt_bf16_kernel<<<1024, 256, 0, stream>>>(Wo, wo16, Dc * Dc);

  // 2) trajectory similarity matrix
  traj_kernel<<<Nc, 256, 0, stream>>>(tpos, tquat, traj);

  // 3) Q/K/V projections (WMMA + TDM-staged A/W tiles)
  dim3 gg(Mrows / 64, Dc / 128);
  gemm_bf16_kernel<0><<<gg, 256, 0, stream>>>(data16, wq16, bq, Qp);
  gemm_bf16_kernel<0><<<gg, 256, 0, stream>>>(data16, wk16, bk, Kp);
  gemm_bf16_kernel<1><<<gg, 256, 0, stream>>>(data16, wv16, bv, Vt);

  // 4) fused attention + entropy/certainty
  attn_kernel<<<Bc * (Nc / 16), 128, 0, stream>>>(Qp, Kp, Vt, traj, cert, aout, cert_o);

  // 5) output projection -> d_out (f32)
  gemm_bf16_kernel<2><<<gg, 256, 0, stream>>>(aout, wo16, bo, out_f);
}
